// BoundaryPredictor1_87067577025311
// MI455X (gfx1250) — matrix operand order repaired and checked by reading, not compiled, not run on hardware
//
#include <hip/hip_runtime.h>
#include <hip/hip_bf16.h>

// Problem constants (match reference)
constexpr int kB = 8;
constexpr int kL = 2048;
constexpr int kD = 512;
constexpr int kH = 2048;
constexpr float kEPS = 1e-6f;

// Output layout (flat f32): pooled[B,L,D] | loss | hb.sum | n.sum | short_mask[B,L]
constexpr long kPooledOff = 0;
constexpr long kLossOff   = (long)kB * kL * kD;        // 8,388,608
constexpr long kHbSumOff  = kLossOff + 1;
constexpr long kNSumOff   = kLossOff + 2;
constexpr long kSMOff     = kLossOff + 3;

typedef __attribute__((ext_vector_type(8)))  _Float16 h8_t;
typedef __attribute__((ext_vector_type(16))) _Float16 h16_t;
typedef __attribute__((ext_vector_type(8)))  float    v8f;

// ---------------------------------------------------------------------------
// Kernel 0: zero the 3 scalar accumulator slots in d_out
// ---------------------------------------------------------------------------
__global__ void zero3_kernel(float* __restrict__ out) {
    if (threadIdx.x < 3) out[kLossOff + threadIdx.x] = 0.0f;
}

// ---------------------------------------------------------------------------
// Kernel 1: W1 [D,H] f32 (row-major) -> W1^T [H,D] f16 (row-major)
// 32x32 LDS tile transpose, block (32,8), grid (H/32, D/32)
// ---------------------------------------------------------------------------
__global__ __launch_bounds__(256) void transpose_w1_kernel(
    const float* __restrict__ W1, _Float16* __restrict__ w1t) {
    __shared__ float tile[32][33];
    const int tx = threadIdx.x, ty = threadIdx.y;
    const int h0 = blockIdx.x * 32;
    const int d0 = blockIdx.y * 32;
    #pragma unroll
    for (int j = 0; j < 32; j += 8)
        tile[ty + j][tx] = W1[(long)(d0 + ty + j) * kH + (h0 + tx)];
    __syncthreads();
    #pragma unroll
    for (int j = 0; j < 32; j += 8)
        w1t[(long)(h0 + ty + j) * kD + (d0 + tx)] = (_Float16)tile[tx][ty + j];
}

// ---------------------------------------------------------------------------
// Kernel 2: fused boundary-MLP GEMM.
// logits[m] = sum_n relu( sum_k hidden[m,k]*W1[k,n] + b1[n] ) * W2[n] + b2
// Block = 128 threads (4 waves), 64 token rows per block (four 16-row M-tiles
// per wave) so each streamed B fragment feeds FOUR WMMAs -> W1^T L2 traffic
// drops to 256 blocks x 2 MB = 0.5 GB; GEMM becomes compute-bound.
// WMMA f32_16x16x32_f16; A from LDS (f16 tile), B streamed from W1^T (L2)
// with a software prefetch of the next N-chunk.
// ---------------------------------------------------------------------------
__global__ __launch_bounds__(128) void gemm_logits_kernel(
    const float* __restrict__ hidden, const _Float16* __restrict__ w1t,
    const float* __restrict__ b1, const float* __restrict__ w2,
    const float* __restrict__ b2, float* __restrict__ logits) {
    __shared__ _Float16 atile[64 * 520];   // 64 rows x 512 f16, +8 pad/row (~66.5 KB)
    __shared__ float red[4][64];

    const int tid  = threadIdx.x;
    const int lane = tid & 31;
    const int wave = tid >> 5;
    const long tokbase = (long)blockIdx.x * 64;

    // Stage 64x512 f32 hidden tile -> f16 LDS (float4 loads, coalesced)
    for (int idx = tid; idx < 64 * 128; idx += 128) {
        const int row = idx >> 7;      // 0..63
        const int c4  = idx & 127;     // 0..127 (float4 index)
        const float4 v = *(const float4*)(hidden + (tokbase + row) * kD + c4 * 4);
        _Float16* p = &atile[row * 520 + c4 * 4];
        p[0] = (_Float16)v.x; p[1] = (_Float16)v.y;
        p[2] = (_Float16)v.z; p[3] = (_Float16)v.w;
    }
    __syncthreads();

    // A layout (matches B): lanes 0-15 -> row=lane, K 0..15 +base
    //                      lanes 16-31 -> row=lane-16, K 16..31 +base
    const int r     = lane & 15;
    const int koffA = (lane >> 4) * 16;
    // B (32x16): lanes 0-15 -> col=lane, K 0..15; lanes 16-31 -> col, K 16..31
    const int koffB = (lane >> 4) * 16;

    float lacc[4][8];
    #pragma unroll
    for (int j = 0; j < 4; ++j)
        #pragma unroll
        for (int i = 0; i < 8; ++i) lacc[j][i] = 0.0f;

    for (int nc = wave * 16; nc < kH; nc += 64) {
        v8f acc[4];
        #pragma unroll
        for (int j = 0; j < 4; ++j) acc[j] = (v8f){};
        const int n = nc + r;
        const _Float16* bp = w1t + (long)n * kD;
        // Prefetch next N-chunk's B rows while this chunk computes.
        if (nc + 64 < kH) __builtin_prefetch(bp + 64l * kD, 0, 1);
        #pragma unroll 2
        for (int ks = 0; ks < kD; ks += 32) {
            // Shared B fragment for all four M-tiles
            const h16_t bf = *(const h16_t*)(bp + ks + koffB);
            #pragma unroll
            for (int j = 0; j < 4; ++j) {
                const int rowbase = (j * 16 + r) * 520;
                const h8_t alo = *(const h8_t*)(&atile[rowbase + ks + koffA]);
                const h8_t ahi = *(const h8_t*)(&atile[rowbase + ks + 8 + koffA]);
                h16_t a;
                #pragma unroll
                for (int i = 0; i < 8; ++i) { a[i] = alo[i]; a[8 + i] = ahi[i]; }
                acc[j] = __builtin_amdgcn_wmma_f32_16x16x32_f16(
                    false, a, false, bf, (short)0, acc[j], false, false);
            }
        }
        // Fuse relu(+b1) * W2[n]; accumulate over N.
        const float bias = b1[n];
        const float w2n  = w2[n];
        #pragma unroll
        for (int j = 0; j < 4; ++j) {
            #pragma unroll
            for (int i = 0; i < 8; ++i) {
                float v = acc[j][i] + bias;   // acc[j][i]: row = 16j + i (+8 if lane>=16)
                v = v > 0.0f ? v : 0.0f;
                lacc[j][i] += v * w2n;
            }
        }
    }

    // Sum across the 16 lanes that hold the same row (halves stay separate).
    #pragma unroll
    for (int j = 0; j < 4; ++j) {
        #pragma unroll
        for (int i = 0; i < 8; ++i) {
            float v = lacc[j][i];
            v += __shfl_xor(v, 1);
            v += __shfl_xor(v, 2);
            v += __shfl_xor(v, 4);
            v += __shfl_xor(v, 8);
            lacc[j][i] = v;
        }
    }
    if (lane == 0) {
        #pragma unroll
        for (int j = 0; j < 4; ++j)
            #pragma unroll
            for (int i = 0; i < 8; ++i) red[wave][j * 16 + i] = lacc[j][i];       // rows 16j..16j+7
    }
    if (lane == 16) {
        #pragma unroll
        for (int j = 0; j < 4; ++j)
            #pragma unroll
            for (int i = 0; i < 8; ++i) red[wave][j * 16 + 8 + i] = lacc[j][i];   // rows 16j+8..16j+15
    }
    __syncthreads();
    if (tid < 64) {
        const float s = red[0][tid] + red[1][tid] + red[2][tid] + red[3][tid] + b2[0];
        logits[tokbase + tid] = s;
    }
}

// ---------------------------------------------------------------------------
// Kernel 3: per-batch boundary logic, scan, segment table, loss, short_mask.
// One block (256 threads) per batch item; 8 tokens per thread.
// ---------------------------------------------------------------------------
__global__ __launch_bounds__(256) void boundary_kernel(
    const float* __restrict__ logits, const float* __restrict__ mask,
    const float* __restrict__ target, const float* __restrict__ u,
    float* __restrict__ out, int* __restrict__ segstart, int* __restrict__ socc) {
    const int b = blockIdx.x;
    const int t = threadIdx.x;
    __shared__ float sc[256];
    __shared__ float lastv[256];
    __shared__ float sh_len;

    const int base = b * kL + t * 8;
    float hb[8];
    float msum = 0.0f;
    #pragma unroll
    for (int i = 0; i < 8; ++i) {
        const float lg = logits[base + i];
        const float uu = u[base + i];
        const float noise = logf(uu) - log1pf(-uu);
        const float x = lg + noise;                 // TEMP = 1
        const float soft = 1.0f / (1.0f + expf(-x));
        const float hard = soft > 0.5f ? 1.0f : 0.0f;
        float v = (hard - soft) + soft;             // straight-through fwd (== hard)
        const float m = mask[base + i];
        v *= m;
        hb[i] = v;
        msum += m;
    }

    // valid length = sum(mask)
    sc[t] = msum; __syncthreads();
    for (int off = 128; off > 0; off >>= 1) {
        if (t < off) sc[t] += sc[t + off];
        __syncthreads();
    }
    if (t == 0) sh_len = sc[0];
    __syncthreads();
    const float nf = sh_len;
    const int len = (int)(nf + 0.5f);

    // Force a boundary at the last real token of padded rows.
    if (len < kL) {
        const int lr = len - 1;                     // len >= L/2 >= 1
        if (lr >= t * 8 && lr < t * 8 + 8) hb[lr - t * 8] = fmaxf(hb[lr - t * 8], 1.0f);
    }

    float s = 0.0f;
    #pragma unroll
    for (int i = 0; i < 8; ++i) s += hb[i];
    lastv[t] = hb[7];
    __syncthreads();

    // Block inclusive scan of per-thread sums (Hillis-Steele).
    sc[t] = s; __syncthreads();
    for (int off = 1; off < 256; off <<= 1) {
        const float v = (t >= off) ? sc[t - off] : 0.0f;
        __syncthreads();
        sc[t] += v;
        __syncthreads();
    }
    const float excl  = sc[t] - s;
    const float total = sc[255];                    // counts[b]

    // Segment ids (exclusive cumsum), segment-start table, short_mask.
    float running = excl;
    #pragma unroll
    for (int i = 0; i < 8; ++i) {
        const int l = t * 8 + i;
        const float prevhb = (i == 0) ? ((t == 0) ? 1.0f : lastv[t - 1]) : hb[i - 1];
        const int seg = (int)(running + 0.5f);
        if (prevhb >= 0.5f) segstart[b * (kL + 1) + seg] = l;
        out[kSMOff + (long)b * kL + l] = ((float)l < total) ? 1.0f : 0.0f;
        running += hb[i];
    }
    __syncthreads();

    if (t == 0) {
        const float hbLast = lastv[255];
        const int S = (int)(total - hbLast + 0.5f) + 1;   // seg[L-1] + 1
        segstart[b * (kL + 1) + S] = kL;
        socc[b] = S;
        // Binomial NLL vs target boundary counts
        float p = target[b] / nf;
        p = fminf(fmaxf(p, kEPS), 1.0f - kEPS);
        const float c = total;
        const float logpmf = lgammaf(nf + 1.0f) - lgammaf(c + 1.0f)
                           - lgammaf(nf - c + 1.0f)
                           + c * logf(p) + (nf - c) * log1pf(-p);
        atomicAdd(out + kLossOff, -logpmf / (float)kB);
        atomicAdd(out + kHbSumOff, c);
        atomicAdd(out + kNSumOff, nf);
    }
}

// ---------------------------------------------------------------------------
// Kernel 4: segment mean-pool. One block per (b, s); 128 threads x float4 = 512 d.
// ---------------------------------------------------------------------------
__global__ __launch_bounds__(128) void pool_kernel(
    const float* __restrict__ hidden, const int* __restrict__ segstart,
    const int* __restrict__ socc, float* __restrict__ out) {
    const int blk  = blockIdx.x;
    const int b    = blk >> 11;     // / 2048
    const int sidx = blk & 2047;
    const int t    = threadIdx.x;

    float4 acc = make_float4(0.f, 0.f, 0.f, 0.f);
    float inv = 0.0f;
    if (sidx < socc[b]) {
        const int st = segstart[b * (kL + 1) + sidx];
        const int en = segstart[b * (kL + 1) + sidx + 1];
        for (int l = st; l < en; ++l) {
            const float4 v = *(const float4*)(hidden + ((long)(b * kL + l)) * kD + t * 4);
            acc.x += v.x; acc.y += v.y; acc.z += v.z; acc.w += v.w;
        }
        inv = 1.0f / ((float)(en - st) + 1e-9f);
    }
    const float4 o = make_float4(acc.x * inv, acc.y * inv, acc.z * inv, acc.w * inv);
    *(float4*)(out + kPooledOff + ((long)(b * kL + sidx)) * kD + t * 4) = o;
}

// ---------------------------------------------------------------------------
extern "C" void kernel_launch(void* const* d_in, const int* in_sizes, int n_in,
                              void* d_out, int out_size, void* d_ws, size_t ws_size,
                              hipStream_t stream) {
    (void)in_sizes; (void)n_in; (void)out_size; (void)ws_size;
    const float* hidden = (const float*)d_in[0];   // [B,L,D]
    const float* mask   = (const float*)d_in[1];   // [B,L]
    const float* target = (const float*)d_in[2];   // [B]
    const float* W1     = (const float*)d_in[3];   // [D,H]
    const float* b1     = (const float*)d_in[4];   // [H]
    const float* W2     = (const float*)d_in[5];   // [H,1]
    const float* b2     = (const float*)d_in[6];   // [1]
    const float* u      = (const float*)d_in[7];   // [B,L]
    float* out = (float*)d_out;

    // Workspace layout
    char* ws = (char*)d_ws;
    _Float16* w1t   = (_Float16*)ws;                                    // 2 MB
    float*    logit = (float*)(ws + (2u << 20));                        // 64 KB
    int*      segst = (int*)(ws + (2u << 20) + 65536);                  // 8*(L+1)*4 B
    int*      socc  = (int*)(ws + (2u << 20) + 65536 + kB * (kL + 1) * 4);

    zero3_kernel<<<1, 32, 0, stream>>>(out);
    transpose_w1_kernel<<<dim3(kH / 32, kD / 32), dim3(32, 8), 0, stream>>>(W1, w1t);
    gemm_logits_kernel<<<(kB * kL) / 64, 128, 0, stream>>>(hidden, w1t, b1, W2, b2, logit);
    boundary_kernel<<<kB, 256, 0, stream>>>(logit, mask, target, u, out, segst, socc);
    pool_kernel<<<kB * kL, 128, 0, stream>>>(hidden, segst, socc, out);
}
